// _NonLocalBlockND_30820685316716
// MI455X (gfx1250) — matrix operand order repaired and checked by reading, not compiled, not run on hardware
//
#include <hip/hip_runtime.h>
#include <hip/hip_bf16.h>

typedef __attribute__((ext_vector_type(16))) _Float16 v16h;
typedef __attribute__((ext_vector_type(8)))  _Float16 v8h;
typedef __attribute__((ext_vector_type(4)))  _Float16 v4h;
typedef __attribute__((ext_vector_type(8)))  float    v8f;
typedef __attribute__((ext_vector_type(2)))  int      i32x2;
typedef __attribute__((ext_vector_type(4)))  int      i32x4;

#define GAS __attribute__((address_space(1)))   // global
#define LAS __attribute__((address_space(3)))   // LDS

#define C_INCH 256
#define C8CH   32
#define NF     4096   // T*32*32  (full-res spatial per batch)
#define NK     1024   // T*16*16  (pooled spatial per batch)
#define NB     16     // batch
#define EPSBN  1e-5f

// CDNA5 async global->LDS path (guarded: falls back to register staging if absent)
#if defined(__gfx1250__) && \
    __has_builtin(__builtin_amdgcn_global_load_async_to_lds_b128) && \
    __has_builtin(__builtin_amdgcn_global_load_async_to_lds_b64) && \
    __has_builtin(__builtin_amdgcn_s_wait_asynccnt)
#define HAVE_ASYNC_LDS 1
#else
#define HAVE_ASYNC_LDS 0
#endif

enum { EPI_F16 = 0, EPI_F16_T = 1, EPI_F32 = 2, EPI_F16_SCRAMBLE = 3 };

// Block tile: every GEMM here has M%32==0, N%128==0, K%32==0 -> no guards.
#define BM 32
#define BN 128
#define BK 32
#define GT 256   // 8 waves (wave32)

// ---------------------------------------------------------------------------
// Batched WMMA GEMM: C[M,N] = A[M,K] (f16 row-major, ld=lda)
//                           @ B (COLUMN-major: elem[k][n] = Bp[n*ldb + k], f16 or f32)
// f32 accumulate via v_wmma_f32_16x16x32_f16; 2 N-tiles per wave (A-frag reuse);
// double-buffered LDS, one barrier per K-step. f16-B path uses CDNA5 async
// global->LDS copies (ASYNCcnt); f32-B path stages via registers (needs cvt).
// ---------------------------------------------------------------------------
template <int EPI, bool BF32>
__global__ __launch_bounds__(GT)
void gemm_wmma(const _Float16* __restrict__ A, long lda, long sA,
               const void* __restrict__ Bp, long ldb, long sB,
               void* __restrict__ Op, long ldo, long sO,
               const float* __restrict__ bias, float scale,
               int M, int N, int K)
{
    __shared__ _Float16 As[2][BM][BK + 8];   // row stride 80B (16B-aligned)
    __shared__ _Float16 Bs[2][BN][BK + 8];   // column-major B tile: Bs[n][k]

    const int bz = blockIdx.z;
    const _Float16* Ab = A + (long)bz * sA;

    const int mBase = blockIdx.y * BM;
    const int nBase = blockIdx.x * BN;
    const int tid  = threadIdx.x;
    const int lane = tid & 31;
    const int wave = tid >> 5;
    const int lg   = lane >> 4;          // half-wave group (ISA 16-bit layouts)
    const int lm   = lane & 15;
    const int wm   = (wave & 1) * 16;    // 2 M-tiles
    const int wn   = (wave >> 1) * 32;   // 4 wave-columns x 2 N-tiles each

    // branch-free staging coordinates
    const int am = tid >> 3;             // 0..31   (A: 4 halfs/thread)
    const int ak = (tid & 7) * 4;
    const int bn = tid >> 1;             // 0..127  (B: 16 halfs/thread, K-contig)
    const int bk = (tid & 1) * 16;

    const _Float16* aSrc = Ab + (long)(mBase + am) * lda + ak;
    const char* bSrcBase = (const char*)Bp
        + ((long)bz * sB + (long)(nBase + bn) * ldb + bk) * (BF32 ? 4 : 2);

    v8f acc0 = {}, acc1 = {};
    const int iters = K / BK;

    // fragment loads + 2 WMMAs from LDS buffer `buf` (contiguous k order: lane half lg holds k = lg*16 .. +15, same for A and B)
    auto mm = [&](int buf) {
        union { v16h v; v8h h[2]; } af, bf0, bf1;
        af.h[0]  = *(const v8h*)&As[buf][wm + lm][lg * 16];       // K = lg*16 .. +7
        af.h[1]  = *(const v8h*)&As[buf][wm + lm][lg * 16 + 8];   // K = lg*16+8 .. +15
        bf0.h[0] = *(const v8h*)&Bs[buf][wn + lm][lg * 16];       // K = lg*16 .. +7
        bf0.h[1] = *(const v8h*)&Bs[buf][wn + lm][lg * 16 + 8];
        bf1.h[0] = *(const v8h*)&Bs[buf][wn + 16 + lm][lg * 16];
        bf1.h[1] = *(const v8h*)&Bs[buf][wn + 16 + lm][lg * 16 + 8];
        acc0 = __builtin_amdgcn_wmma_f32_16x16x32_f16(false, af.v, false, bf0.v,
                                                      (short)0, acc0, false, false);
        acc1 = __builtin_amdgcn_wmma_f32_16x16x32_f16(false, af.v, false, bf1.v,
                                                      (short)0, acc1, false, false);
    };

#if HAVE_ASYNC_LDS
    if constexpr (!BF32) {
        // ---- async global->LDS pipeline (GLOBAL_LOAD_ASYNC_TO_LDS_*, ASYNCcnt) ----
        const _Float16* bSrcH = (const _Float16*)bSrcBase;
        auto issue = [&](int k0, int buf) {
            __builtin_amdgcn_global_load_async_to_lds_b64(
                (GAS i32x2*)(aSrc + k0),
                (LAS i32x2*)&As[buf][am][ak], 0, 0);
            __builtin_amdgcn_global_load_async_to_lds_b128(
                (GAS i32x4*)(bSrcH + k0),
                (LAS i32x4*)&Bs[buf][bn][bk], 0, 0);
            __builtin_amdgcn_global_load_async_to_lds_b128(
                (GAS i32x4*)(bSrcH + k0 + 8),
                (LAS i32x4*)&Bs[buf][bn][bk + 8], 0, 0);
        };
        issue(0, 0);
        for (int it = 0; it < iters; ++it) {
            const int buf = it & 1;
            __builtin_amdgcn_s_wait_asynccnt(0);   // own tile-it copies landed in LDS
            __syncthreads();                        // all waves' copies visible; buf^1 reads done
            if (it + 1 < iters) issue((it + 1) * BK, buf ^ 1);   // overlap with math
            if (it + 2 < iters)
                __builtin_prefetch((const char*)(bSrcH + (long)(it + 2) * BK), 0, 0);
            mm(buf);
        }
    } else
#endif
    {
        // ---- register-staged pipeline (f32 B needs conversion during staging) ----
        v4h aReg;
        v8h bReg0, bReg1;
        auto loadGlobal = [&](int k0) {
            aReg = *(const v4h*)(aSrc + k0);
            if (BF32) {
                const float* src = (const float*)bSrcBase + k0;
                v8f f0 = *(const v8f*)(src);
                v8f f1 = *(const v8f*)(src + 8);
#pragma unroll
                for (int j = 0; j < 8; ++j) { bReg0[j] = (_Float16)f0[j]; bReg1[j] = (_Float16)f1[j]; }
            } else {
                const _Float16* src = (const _Float16*)bSrcBase + k0;
                bReg0 = *(const v8h*)(src);
                bReg1 = *(const v8h*)(src + 8);
            }
        };
        loadGlobal(0);
        for (int it = 0; it < iters; ++it) {
            const int buf = it & 1;
            *(v4h*)&As[buf][am][ak]     = aReg;
            *(v8h*)&Bs[buf][bn][bk]     = bReg0;
            *(v8h*)&Bs[buf][bn][bk + 8] = bReg1;
            __syncthreads();
            if (it + 1 < iters) loadGlobal((it + 1) * BK);
            if (it + 2 < iters)
                __builtin_prefetch(bSrcBase + (long)(it + 2) * BK * (BF32 ? 4 : 2), 0, 0);
            mm(buf);
        }
    }

    // ---- epilogue: C/D layout = VGPR r -> M = r + lg*8, N = lm ----
#pragma unroll
    for (int t = 0; t < 2; ++t) {
        const v8f a = t ? acc1 : acc0;
        const int gn = nBase + wn + t * 16 + lm;
#pragma unroll
        for (int r = 0; r < 8; ++r) {
            const int gm = mBase + wm + lg * 8 + r;
            float v = a[r] * scale;
            if (bias) v += bias[gm];
            if (EPI == EPI_F16) {
                ((_Float16*)Op)[(long)bz * sO + (long)gm * ldo + gn] = (_Float16)v;
            } else if (EPI == EPI_F16_T) {
                ((_Float16*)Op)[(long)bz * sO + (long)gn * ldo + gm] = (_Float16)v;
            } else if (EPI == EPI_F32) {
                ((float*)Op)[(long)bz * sO + (long)gm * ldo + gn] = v;
            } else { // EPI_F16_SCRAMBLE: fused dim-scrambling reshape, TRANSPOSED out
                int cc = (gm & 63) * 4 + (gn >> 10);
                int n2 = (gm >> 6) * 1024 + (gn & 1023);
                ((_Float16*)Op)[(long)bz * sO + (long)n2 * ldo + cc] = (_Float16)v;
            }
        }
    }
}

// ---------------------------------------------------------------------------
// Elementwise / transform helpers
// ---------------------------------------------------------------------------
__global__ void f32_to_f16(const float* __restrict__ s, _Float16* __restrict__ d, long n)
{
    long i = (long)blockIdx.x * blockDim.x + threadIdx.x;
    if (i < n) d[i] = (_Float16)s[i];
}

// x [NB][256][4096] f32 -> Xh_t [NB][4096][256] f16 (tiled LDS transpose)
__global__ __launch_bounds__(256)
void transpose_f32_to_f16(const float* __restrict__ src, _Float16* __restrict__ dst)
{
    __shared__ _Float16 t[32][33];
    const int b  = blockIdx.z;
    const int c0 = blockIdx.y * 32;
    const int n0 = blockIdx.x * 32;
    const int tx = threadIdx.x & 31, ty = threadIdx.x >> 5;   // 32x8
    const float*   s = src + (long)b * C_INCH * NF;
    _Float16*      d = dst + (long)b * NF * C_INCH;
#pragma unroll
    for (int j = 0; j < 32; j += 8)
        t[ty + j][tx] = (_Float16)s[(long)(c0 + ty + j) * NF + (n0 + tx)];
    __syncthreads();
#pragma unroll
    for (int j = 0; j < 32; j += 8)
        d[(long)(n0 + ty + j) * C_INCH + (c0 + tx)] = t[tx][ty + j];
}

// MaxPool3d(1,2,2): src [NB][C][4][32][32] f16 -> dst [NB][C][1024] or transposed [NB][1024][C]
template <bool TRANS>
__global__ void pool2x2_f16(const _Float16* __restrict__ src, _Float16* __restrict__ dst,
                            int C, long total)
{
    long i = (long)blockIdx.x * blockDim.x + threadIdx.x;
    if (i >= total) return;
    int  s  = (int)(i % NK);
    long bc = i / NK;
    int  c  = (int)(bc % C);
    long b  = bc / C;
    int hp = s & 15, wp = (s >> 4) & 15, t = s >> 8;
    const _Float16* p = src + (b * C + c) * (long)NF + t * 1024 + wp * 64 + hp * 2;
    float a0 = (float)p[0], a1 = (float)p[1], a2 = (float)p[32], a3 = (float)p[33];
    _Float16 m = (_Float16)fmaxf(fmaxf(a0, a1), fmaxf(a2, a3));
    if (TRANS) dst[(b * NK + s) * C + c] = m;
    else       dst[(b * C + c) * NK + s] = m;
}

// In-place softmax over rows of length 1024 (NB*NF rows), f32
__global__ __launch_bounds__(256)
void softmax_rows(float* __restrict__ att)
{
    __shared__ float red[256];
    float* p = att + (long)blockIdx.x * NK;
    int tid = threadIdx.x;
    float v[4], mx = -3.4e38f;
#pragma unroll
    for (int i = 0; i < 4; ++i) { v[i] = p[tid + i * 256]; mx = fmaxf(mx, v[i]); }
    red[tid] = mx; __syncthreads();
    for (int s = 128; s > 0; s >>= 1) { if (tid < s) red[tid] = fmaxf(red[tid], red[tid + s]); __syncthreads(); }
    mx = red[0]; __syncthreads();
    float sum = 0.f;
#pragma unroll
    for (int i = 0; i < 4; ++i) { v[i] = __expf(v[i] - mx); sum += v[i]; }
    red[tid] = sum; __syncthreads();
    for (int s = 128; s > 0; s >>= 1) { if (tid < s) red[tid] += red[tid + s]; __syncthreads(); }
    float inv = 1.f / red[0];
#pragma unroll
    for (int i = 0; i < 4; ++i) p[tid + i * 256] = v[i] * inv;
}

// Per-channel batch stats (biased var) over [NB][256][NF] f32; one block per channel
__global__ __launch_bounds__(256)
void bn_stats(const float* __restrict__ Z, float* __restrict__ mean, float* __restrict__ rstd)
{
    __shared__ float s1[256], s2[256];
    int c = blockIdx.x, tid = threadIdx.x;
    float sum = 0.f, sq = 0.f;
    for (int i = tid; i < NB * NF; i += 256) {
        int bb = i >> 12, n = i & (NF - 1);
        float v = Z[(long)bb * (C_INCH * (long)NF) + (long)c * NF + n];
        sum += v; sq += v * v;
    }
    s1[tid] = sum; s2[tid] = sq; __syncthreads();
    for (int s = 128; s > 0; s >>= 1) {
        if (tid < s) { s1[tid] += s1[tid + s]; s2[tid] += s2[tid + s]; }
        __syncthreads();
    }
    if (tid == 0) {
        float m   = s1[0] * (1.f / (NB * NF));
        float var = s2[0] * (1.f / (NB * NF)) - m * m;
        mean[c] = m;
        rstd[c] = rsqrtf(var + EPSBN);
    }
}

__global__ void bn_combine(const float* __restrict__ Z1, const float* __restrict__ Z2,
                           const float* __restrict__ m1, const float* __restrict__ r1,
                           const float* __restrict__ m2, const float* __restrict__ r2,
                           const float* __restrict__ gw, const float* __restrict__ bw,
                           const float* __restrict__ gi, const float* __restrict__ bi,
                           float* __restrict__ out, long total)
{
    long i = (long)blockIdx.x * blockDim.x + threadIdx.x;
    if (i >= total) return;
    int c = (int)((i >> 12) & 255);
    out[i] = gw[c] * (Z1[i] - m1[c]) * r1[c] + bw[c]
           + gi[c] * (Z2[i] - m2[c]) * r2[c] + bi[c];
}

// ---------------------------------------------------------------------------
extern "C" void kernel_launch(void* const* d_in, const int* in_sizes, int n_in,
                              void* d_out, int out_size, void* d_ws, size_t ws_size,
                              hipStream_t stream)
{
    const float* x    = (const float*)d_in[0];
    const float* Wq   = (const float*)d_in[1];
    const float* bq   = (const float*)d_in[2];
    const float* Wk   = (const float*)d_in[3];
    const float* bk   = (const float*)d_in[4];
    const float* Wv   = (const float*)d_in[5];
    const float* bv   = (const float*)d_in[6];
    const float* Ww   = (const float*)d_in[7];
    const float* bw   = (const float*)d_in[8];
    const float* gw   = (const float*)d_in[9];
    const float* betw = (const float*)d_in[10];
    const float* Win  = (const float*)d_in[11];
    const float* b_in = (const float*)d_in[12];
    const float* gin  = (const float*)d_in[13];
    const float* beti = (const float*)d_in[14];

    float* out_t = (float*)d_out;                      // [16,256,4,32,32]
    float* att   = out_t + (long)NB * C_INCH * NF;     // [16,4096,1024]

    char* wsp = (char*)d_ws;
    auto alloc = [&](size_t bytes) -> char* {
        char* p = wsp; wsp += (bytes + 255) & ~(size_t)255; return p;
    };
    _Float16* Xh_t = (_Float16*)alloc((size_t)NB * NF * C_INCH * 2);  // [b][n][c]
    _Float16* Wqh  = (_Float16*)alloc((size_t)C8CH * C_INCH * 2);
    _Float16* Wkh  = (_Float16*)alloc((size_t)C8CH * C_INCH * 2);
    _Float16* Wvh  = (_Float16*)alloc((size_t)C_INCH * C_INCH * 2);
    _Float16* Wwh  = (_Float16*)alloc((size_t)C_INCH * C_INCH * 2);
    _Float16* Winh = (_Float16*)alloc((size_t)C_INCH * C_INCH * 2);
    _Float16* Qh   = (_Float16*)alloc((size_t)NB * NF * C8CH * 2);    // [b][4096][32]
    _Float16* Kf   = (_Float16*)alloc((size_t)NB * C8CH * NF * 2);    // pre-pool [b][32][4096]
    _Float16* Vf   = (_Float16*)alloc((size_t)NB * C_INCH * NF * 2);  // pre-pool [b][256][4096]
    _Float16* Kp_t = (_Float16*)alloc((size_t)NB * NK * C8CH * 2);    // [b][1024][32]
    _Float16* Vp   = (_Float16*)alloc((size_t)NB * C_INCH * NK * 2);  // [b][256][1024]
    _Float16* Yh_t = (_Float16*)alloc((size_t)NB * NF * C_INCH * 2);  // [b][4096][256] scrambled^T
    float*    Z1   = (float*)alloc((size_t)NB * C_INCH * NF * 4);
    float*    Z2   = (float*)alloc((size_t)NB * C_INCH * NF * 4);
    float*    m1   = (float*)alloc(256 * 4);
    float*    r1   = (float*)alloc(256 * 4);
    float*    m2   = (float*)alloc(256 * 4);
    float*    r2   = (float*)alloc(256 * 4);

    dim3 blk(GT);
    auto g1d = [](long n) { return dim3((unsigned)((n + 255) / 256)); };
    const long sX = (long)NF * C_INCH;   // per-batch stride of Xh_t / Yh_t / Vf / Z*

    // 0) conversions: x -> transposed f16 (K-contiguous B operand); weights -> f16
    transpose_f32_to_f16<<<dim3(NF / 32, C_INCH / 32, NB), 256, 0, stream>>>(x, Xh_t);
    f32_to_f16<<<g1d(C8CH * C_INCH), 256, 0, stream>>>(Wq, Wqh, C8CH * C_INCH);
    f32_to_f16<<<g1d(C8CH * C_INCH), 256, 0, stream>>>(Wk, Wkh, C8CH * C_INCH);
    f32_to_f16<<<g1d(C_INCH * C_INCH), 256, 0, stream>>>(Wv, Wvh, C_INCH * C_INCH);
    f32_to_f16<<<g1d(C_INCH * C_INCH), 256, 0, stream>>>(Ww, Wwh, C_INCH * C_INCH);
    f32_to_f16<<<g1d(C_INCH * C_INCH), 256, 0, stream>>>(Win, Winh, C_INCH * C_INCH);

    // 1) q = (Wq@X)^T + bq -> Qh [b][4096][32]
    gemm_wmma<EPI_F16_T, false><<<dim3(NF / BN, C8CH / BM, NB), blk, 0, stream>>>(
        Wqh, C_INCH, 0, Xh_t, C_INCH, sX, Qh, C8CH, (long)NF * C8CH, bq, 1.f, C8CH, NF, C_INCH);
    // 2) k_full = Wk@X + bk -> Kf [b][32][4096]
    gemm_wmma<EPI_F16, false><<<dim3(NF / BN, C8CH / BM, NB), blk, 0, stream>>>(
        Wkh, C_INCH, 0, Xh_t, C_INCH, sX, Kf, NF, (long)C8CH * NF, bk, 1.f, C8CH, NF, C_INCH);
    // 3) v_full = Wv@X + bv -> Vf [b][256][4096]
    gemm_wmma<EPI_F16, false><<<dim3(NF / BN, C_INCH / BM, NB), blk, 0, stream>>>(
        Wvh, C_INCH, 0, Xh_t, C_INCH, sX, Vf, NF, sX, bv, 1.f, C_INCH, NF, C_INCH);
    // 4) Z2 = Win@X + b_in (f32)
    gemm_wmma<EPI_F32, false><<<dim3(NF / BN, C_INCH / BM, NB), blk, 0, stream>>>(
        Winh, C_INCH, 0, Xh_t, C_INCH, sX, Z2, NF, sX, b_in, 1.f, C_INCH, NF, C_INCH);

    // 5) pooling: k -> transposed (K-contiguous for energy B), v -> row-major (A operand)
    pool2x2_f16<true ><<<g1d((long)NB * C8CH * NK), 256, 0, stream>>>(Kf, Kp_t, C8CH, (long)NB * C8CH * NK);
    pool2x2_f16<false><<<g1d((long)NB * C_INCH * NK), 256, 0, stream>>>(Vf, Vp, C_INCH, (long)NB * C_INCH * NK);

    // 6) energy S = Qh @ Kp / 64 -> att region (f32). Mean-correction terms are
    //    constant per row -> softmax identical, so they are skipped.
    gemm_wmma<EPI_F32, false><<<dim3(NK / BN, NF / BM, NB), blk, 0, stream>>>(
        Qh, C8CH, (long)NF * C8CH, Kp_t, C8CH, (long)NK * C8CH,
        att, NK, (long)NF * NK, nullptr, 0.015625f, NF, NK, C8CH);

    // 7) softmax rows in place -> attention output
    softmax_rows<<<dim3(NB * NF), 256, 0, stream>>>(att);

    // 8) out = Vp @ att^T; att^T is naturally K-contiguous (ldb = 1024, f32 source
    //    converted during LDS staging); epilogue fuses scramble + transpose -> Yh_t
    gemm_wmma<EPI_F16_SCRAMBLE, true><<<dim3(NF / BN, C_INCH / BM, NB), blk, 0, stream>>>(
        Vp, NK, (long)C_INCH * NK, att, NK, (long)NF * NK,
        Yh_t, C_INCH, sX, nullptr, 1.f, C_INCH, NF, NK);

    // 9) Z1 = Ww@Y + bw (f32)
    gemm_wmma<EPI_F32, false><<<dim3(NF / BN, C_INCH / BM, NB), blk, 0, stream>>>(
        Wwh, C_INCH, 0, Yh_t, C_INCH, sX, Z1, NF, sX, bw, 1.f, C_INCH, NF, C_INCH);

    // 10) BN stats for both branches
    bn_stats<<<dim3(256), 256, 0, stream>>>(Z1, m1, r1);
    bn_stats<<<dim3(256), 256, 0, stream>>>(Z2, m2, r2);

    // 11) final combine -> d_out first region
    bn_combine<<<g1d((long)NB * C_INCH * NF), 256, 0, stream>>>(
        Z1, Z2, m1, r1, m2, r2, gw, betw, gin, beti, out_t, (long)NB * C_INCH * NF);
}
